// EdgeUpdate_19593640804837
// MI455X (gfx1250) — compile-verified, matrix operand order checked by reading, not run on hardware
//
#include <hip/hip_runtime.h>

typedef unsigned short u16;
typedef unsigned int   u32;
typedef __bf16 bf16_t;
typedef bf16_t v16bf __attribute__((ext_vector_type(16)));
typedef float  v8f   __attribute__((ext_vector_type(8)));

#define ATOM_F 64
#define NBR_F  64
#define DIN    192
#define RBF_K  16
#define TILES  4          // 16-edge tiles per wave; block = 8 waves * 4 tiles * 16 = 512 edges

union Frag { uint4 u[2]; v16bf v; };

__device__ __forceinline__ u16 f2bf(float x) {        // hardware RTNE convert
  union { bf16_t h; u16 u; } c; c.h = (bf16_t)x; return c.u;
}
__device__ __forceinline__ u32 pack2(float a, float b) {
  union { bf16_t h[2]; u32 u; } p;
  p.h[0] = (bf16_t)a; p.h[1] = (bf16_t)b;
  return p.u;
}
__device__ __forceinline__ void st4bf(u16* dst, float4 v) {
  uint2 p; p.x = pack2(v.x, v.y); p.y = pack2(v.z, v.w);
  *(uint2*)dst = p;
}
__device__ __forceinline__ float fsigmoid(float x) {
  return __builtin_amdgcn_rcpf(1.0f + __expf(-x));    // v_exp_f32 + v_rcp_f32
}
__device__ __forceinline__ float fsilu(float x) { return x * fsigmoid(x); }

// B fragment: packed contiguously in LDS (32 lanes x 16 bf16 per fragment)
__device__ __forceinline__ v16bf ld_bfrag(const u16* base, int frag, int lane) {
  Frag f;
  const uint4* p = (const uint4*)(base + (frag * 32 + lane) * 16);
  f.u[0] = p[0]; f.u[1] = p[1];
  return f.v;
}
// A fragment from a row-major bf16 tile. ISA 16-bit A layout:
// lane half 0: K = k0+{0..7, 16..23}; lane half 1: K = k0+{8..15, 24..31}
__device__ __forceinline__ v16bf ld_afrag(const u16* tile, int rowElems, int m, int half, int k0) {
  Frag f;
  const u16* row = tile + m * rowElems + k0 + half * 8;
  f.u[0] = *(const uint4*)row;
  f.u[1] = *(const uint4*)(row + 16);
  return f.v;
}

// Repack a (K x 64) f32 weight into bf16 B-fragment order:
// idx = ((kc*4 + nc)*32 + lane)*16 + t ;  k = kc*32 + kl(t,half) ; n = nc*16 + (lane&15)
__device__ __forceinline__ void repack_w(const float* __restrict__ W, u16* dst,
                                         int totalElems, int tid) {
  for (int idx = tid; idx < totalElems; idx += 256) {
    int t  = idx & 15;
    int L  = (idx >> 4) & 31;
    int nc = (idx >> 9) & 3;
    int kc = idx >> 11;
    int half = L >> 4;
    int kl = (t < 8) ? (half * 8 + t) : (16 + half * 8 + (t - 8));
    int k  = kc * 32 + kl;
    int n  = nc * 16 + (L & 15);
    dst[idx] = f2bf(W[k * 64 + n]);
  }
}

__global__ void __launch_bounds__(256)
edge_update_kernel(const float* __restrict__ atom_fea, const float* __restrict__ edge_ij,
                   const int* __restrict__ nbr_atoms, const float* __restrict__ bonds_r,
                   const float* __restrict__ W1, const float* __restrict__ b1,
                   const float* __restrict__ W2, const float* __restrict__ b2,
                   const float* __restrict__ Wr, const float* __restrict__ br,
                   const float* __restrict__ W3, const float* __restrict__ b3,
                   float* __restrict__ out, int E)
{
  extern __shared__ char smem[];
  u16*   w1f     = (u16*)smem;               // 6 KC * 4 NC * 512 = 12288 u16
  u16*   w2f     = w1f + 12288;              // 12288 u16
  u16*   w3f     = w2f + 12288;              // 2*4*512 = 4096 u16
  u16*   wrf     = w3f + 4096;               // 1*4*512 = 2048 u16 (K padded to 32)
  float* biasLds = (float*)(wrf + 2048);     // 256 f32: b1|b2|b3|br
  u16*   xBase   = (u16*)(biasLds + 256);    // 8 waves * 16*192
  u16*   hBase   = xBase + 8 * (16 * DIN);   // 8 waves * 16*64
  u16*   rBase   = hBase + 8 * (16 * NBR_F); // 8 waves * 16*32

  const int tid = threadIdx.x;

  // ---- Phase 0: stage weights (bf16, fragment order) + biases into LDS ----
  repack_w(W1, w1f, 12288, tid);
  repack_w(W2, w2f, 12288, tid);
  repack_w(W3, w3f, 4096, tid);
  for (int idx = tid; idx < 2048; idx += 256) {        // Wr, zero-padded K 16->32
    int t = idx & 15, L = (idx >> 4) & 31, nc = (idx >> 9) & 3;
    int half = L >> 4;
    int kl = (t < 8) ? (half * 8 + t) : (16 + half * 8 + (t - 8));
    int n  = nc * 16 + (L & 15);
    wrf[idx] = (kl < RBF_K) ? f2bf(Wr[kl * 64 + n]) : (u16)0;
  }
  if (tid < 64) {
    biasLds[tid]       = b1[tid];
    biasLds[64 + tid]  = b2[tid];
    biasLds[128 + tid] = b3[tid];
    biasLds[192 + tid] = br[tid];
  }
  __syncthreads();

  const int  wave = tid >> 5, lane = tid & 31;
  u16* xT = xBase + wave * (16 * DIN);
  u16* hT = hBase + wave * (16 * NBR_F);
  u16* rT = rBase + wave * (16 * 32);
  const int m = lane & 15, half = lane >> 4;
  const long waveBase = (long)blockIdx.x * (128 * TILES) + wave * (16 * TILES);

  for (int tIdx = 0; tIdx < TILES; ++tIdx) {
    const long ebase = waveBase + (long)tIdx * 16;

    // ---- Gather: x = [a_i | a_j | edge_ij] as bf16 row-major [16][192] ----
    {
      int r = lane >> 1, hh = lane & 1;                // 2 lanes per edge row
      long e = ebase + r; if (e >= E) e = E - 1;
      int aidx = nbr_atoms[e * 2 + hh];                // hh=0 -> a_i, hh=1 -> a_j
      const float4* asrc = (const float4*)(atom_fea + (long)aidx * ATOM_F);
      u16* xrow = xT + r * DIN;
#pragma unroll
      for (int c = 0; c < 16; ++c) st4bf(xrow + hh * 64 + c * 4, asrc[c]);
      const float4* esrc = (const float4*)(edge_ij + e * NBR_F + hh * 32);
#pragma unroll
      for (int c = 0; c < 8; ++c) st4bf(xrow + 128 + hh * 32 + c * 4, esrc[c]);
      // prefetch next tile's edge row (global_prefetch_b8)
      long en = ebase + 16 + r; if (en >= E) en = E - 1;
      __builtin_prefetch(edge_ij + en * NBR_F + hh * 32, 0, 3);
    }
    if (lane < 16) {                                   // bonds_r tile [16][32], upper K zero
      long e = ebase + lane; if (e >= E) e = E - 1;
      const float4* bsrc = (const float4*)(bonds_r + e * RBF_K);
      u16* rrow = rT + lane * 32;
#pragma unroll
      for (int c = 0; c < 4; ++c) st4bf(rrow + c * 4, bsrc[c]);
      uint4 z = make_uint4(0u, 0u, 0u, 0u);
      *(uint4*)(rrow + 16) = z;
      *(uint4*)(rrow + 24) = z;
      long en = ebase + 16 + lane; if (en >= E) en = E - 1;
      __builtin_prefetch(bonds_r + en * RBF_K, 0, 3);
    }

    // ---- GEMM 1: y1 = x@W1, y2 = x@W2 ; h = silu(y1+b1)*sigmoid(y2+b2) ----
    v16bf xa[6];
#pragma unroll
    for (int kc = 0; kc < 6; ++kc) xa[kc] = ld_afrag(xT, DIN, m, half, kc * 32);
    v16bf ra = ld_afrag(rT, 32, m, half, 0);

#pragma unroll
    for (int nc = 0; nc < 4; ++nc) {
      v8f acc1 = {}, acc2 = {};                        // inline C=0, no splat movs
#pragma unroll
      for (int kc = 0; kc < 6; ++kc) {
        v16bf bw1 = ld_bfrag(w1f, kc * 4 + nc, lane);
        acc1 = __builtin_amdgcn_wmma_f32_16x16x32_bf16(false, xa[kc], false, bw1,
                                                       (short)0, acc1, false, false);
        v16bf bw2 = ld_bfrag(w2f, kc * 4 + nc, lane);
        acc2 = __builtin_amdgcn_wmma_f32_16x16x32_bf16(false, xa[kc], false, bw2,
                                                       (short)0, acc2, false, false);
      }
      float b1v = biasLds[nc * 16 + m];
      float b2v = biasLds[64 + nc * 16 + m];
#pragma unroll
      for (int i = 0; i < 8; ++i) {                    // C layout: row = half*8+i, col = m
        float hv = fsilu(acc1[i] + b1v) * fsigmoid(acc2[i] + b2v);
        int mm = half * 8 + i;
        hT[mm * NBR_F + nc * 16 + m] = f2bf(hv);
      }
    }

    // ---- GEMM 2: out = silu(h@W3+b3) * (bonds_r@Wr+br) ----
    v16bf ha0 = ld_afrag(hT, NBR_F, m, half, 0);
    v16bf ha1 = ld_afrag(hT, NBR_F, m, half, 32);

#pragma unroll
    for (int nc = 0; nc < 4; ++nc) {
      v8f acc3 = {}, accR = {};
      acc3 = __builtin_amdgcn_wmma_f32_16x16x32_bf16(false, ha0, false, ld_bfrag(w3f, nc, lane),
                                                     (short)0, acc3, false, false);
      acc3 = __builtin_amdgcn_wmma_f32_16x16x32_bf16(false, ha1, false, ld_bfrag(w3f, 4 + nc, lane),
                                                     (short)0, acc3, false, false);
      accR = __builtin_amdgcn_wmma_f32_16x16x32_bf16(false, ra, false, ld_bfrag(wrf, nc, lane),
                                                     (short)0, accR, false, false);
      float b3v = biasLds[128 + nc * 16 + m];
      float brv = biasLds[192 + nc * 16 + m];
#pragma unroll
      for (int i = 0; i < 8; ++i) {
        int mm = half * 8 + i;
        long e = ebase + mm;
        if (e < E) out[e * NBR_F + nc * 16 + m] = fsilu(acc3[i] + b3v) * (accR[i] + brv);
      }
    }
  }
}

extern "C" void kernel_launch(void* const* d_in, const int* in_sizes, int n_in,
                              void* d_out, int out_size, void* d_ws, size_t ws_size,
                              hipStream_t stream) {
  const float* atom_fea = (const float*)d_in[0];
  const float* edge_ij  = (const float*)d_in[1];
  const int*   nbr      = (const int*)d_in[2];
  const float* bonds_r  = (const float*)d_in[3];
  const float* W1 = (const float*)d_in[4];
  const float* b1 = (const float*)d_in[5];
  const float* W2 = (const float*)d_in[6];
  const float* b2 = (const float*)d_in[7];
  const float* Wr = (const float*)d_in[8];
  const float* br = (const float*)d_in[9];
  const float* W3 = (const float*)d_in[10];
  const float* b3 = (const float*)d_in[11];
  float* out = (float*)d_out;

  int E = in_sizes[2] / 2;                       // nbr_atoms is (E, 2)
  int edgesPerBlock = 128 * TILES;               // 512
  int blocks = (E + edgesPerBlock - 1) / edgesPerBlock;
  size_t lds = 136192;                           // ~133 KB dynamic LDS (2 blocks/WGP)

  hipFuncSetAttribute((const void*)edge_update_kernel,
                      hipFuncAttributeMaxDynamicSharedMemorySize, (int)lds);

  edge_update_kernel<<<blocks, 256, lds, stream>>>(atom_fea, edge_ij, nbr, bonds_r,
                                                   W1, b1, W2, b2, Wr, br, W3, b3,
                                                   out, E);
}
